// L1AttnSparse_19713899888734
// MI455X (gfx1250) — matrix operand order repaired and checked by reading, not compiled, not run on hardware
//
#include <hip/hip_runtime.h>
#include <hip/hip_bf16.h>
#include <math.h>

// ---------------------------------------------------------------------------
// L1 sliding-window attention for MI455X (gfx1250, wave32, WMMA).
//
// Shapes are fixed by the reference setup: bs=2, n_tok=2048, n_heads=4,
// width=64, WIN=64, src = (dst + j - 32) mod n_tok, all 64 slots populated.
//
// Strategy: per block, one (b, h, 16-dst tile). L1 distances + softmax on
// VALU (they are not a matmul), the PV weighted-sum as a banded 16x96 x 96x64
// f16 WMMA matmul (window union of 16 consecutive dsts spans 79 tokens,
// padded to 96 = 3 K-tiles of 32).
// ---------------------------------------------------------------------------

typedef _Float16 v16h __attribute__((ext_vector_type(16)));
typedef _Float16 v8h  __attribute__((ext_vector_type(8)));
typedef float    v8f  __attribute__((ext_vector_type(8)));

#define NTOK   2048
#define NH     4
#define WID    64
#define WIN    64
#define HALFW  32
#define DT     16            // dst tile
#define SPAN   96            // 79-token window padded to 3 K-tiles of 32
#define NKT    3             // SPAN / 32

// padded LDS strides (floats / halves) to spread bank usage
#define QSTR   68            // 16x64 f32 Q tile
#define KSTR   68            // 80x64 f32 K window
#define VSTR   112           // 64(width) x 96(k) f16, K-major (transposed V)
#define ASTR   65            // 16x64 f32 attn scratch
#define WSTR   104           // 16x96 f16 banded weight matrix

__global__ __launch_bounds__(128) void
l1attn_win_kernel(const float* __restrict__ vg,
                  const float* __restrict__ qg,
                  const float* __restrict__ kg,
                  float* __restrict__ out)
{
    __shared__ __align__(16) float    qs  [DT * QSTR];        //  4352 B
    __shared__ __align__(16) float    ks  [80 * KSTR];        // 21760 B
    __shared__ __align__(32) _Float16 vsT [WID * VSTR];       // 14336 B
    __shared__ __align__(16) float    attn[DT * ASTR];        //  4160 B
    __shared__ __align__(16) _Float16 wmat[DT * WSTR];        //  3328 B

    const int tid  = threadIdx.x;
    const int blk  = blockIdx.x;
    const int tile = blk & 127;          // n_tok/16 = 128 tiles
    const int bh   = blk >> 7;
    const int hh   = bh & 3;
    const int bb   = bh >> 2;
    const int d0   = tile << 4;

    // ---------------- phase 0: zero W and V^T; stage Q and K window --------
    for (int i = tid; i < (DT * WSTR) / 2; i += 128)
        reinterpret_cast<unsigned*>(wmat)[i] = 0u;
    for (int i = tid; i < (WID * VSTR) / 2; i += 128)
        reinterpret_cast<unsigned*>(vsT)[i] = 0u;

    for (int i = tid; i < DT * (WID / 4); i += 128) {          // 256 float4
        const int row = i >> 4, c = (i & 15) << 2;
        const float4 qv = *reinterpret_cast<const float4*>(
            &qg[(((size_t)bb * NTOK + d0 + row) * NH + hh) * WID + c]);
        *reinterpret_cast<float4*>(&qs[row * QSTR + c]) = qv;
    }
    for (int i = tid; i < 80 * (WID / 4); i += 128) {          // 1280 float4
        const int row = i >> 4, c = (i & 15) << 2;
        const int tok = (d0 - HALFW + row) & (NTOK - 1);
        const size_t base = (((size_t)bb * NTOK + tok) * NH + hh) * WID + c;
        const float4 kv = *reinterpret_cast<const float4*>(&kg[base]);
        *reinterpret_cast<float4*>(&ks[row * KSTR + c]) = kv;
        __builtin_prefetch(&vg[base], 0, 0);                   // global_prefetch_b8
    }
    __syncthreads();

    // ---------------- phase 1: V -> f16 V^T in LDS; L1 distances -----------
    for (int i = tid; i < 80 * (WID / 4); i += 128) {
        const int row = i >> 4, c = (i & 15) << 2;             // row = local src s
        const int tok = (d0 - HALFW + row) & (NTOK - 1);
        const float4 vv = *reinterpret_cast<const float4*>(
            &vg[(((size_t)bb * NTOK + tok) * NH + hh) * WID + c]);
        vsT[(c + 0) * VSTR + row] = (_Float16)vv.x;
        vsT[(c + 1) * VSTR + row] = (_Float16)vv.y;
        vsT[(c + 2) * VSTR + row] = (_Float16)vv.z;
        vsT[(c + 3) * VSTR + row] = (_Float16)vv.w;
    }

    // 1024 (dst, j) pairs; threads sharing a dst read the same Q row (LDS bcast)
    #pragma unroll
    for (int it = 0; it < 8; ++it) {
        const int p   = tid + (it << 7);
        const int dst = p >> 6;
        const int j   = p & 63;
        const float4* q4 = reinterpret_cast<const float4*>(&qs[dst * QSTR]);
        const float4* k4 = reinterpret_cast<const float4*>(&ks[(dst + j) * KSTR]);
        float acc = 0.0f;
        #pragma unroll
        for (int c = 0; c < WID / 4; ++c) {
            const float4 qv = q4[c], kv = k4[c];
            acc += fabsf(qv.x - kv.x) + fabsf(qv.y - kv.y)
                 + fabsf(qv.z - kv.z) + fabsf(qv.w - kv.w);
        }
        attn[dst * ASTR + j] = acc * -0.125f;                  // -1/sqrt(64)
    }
    __syncthreads();

    // ---------------- phase 2: softmax + banded f16 weight scatter ---------
    if (tid < DT) {
        const int row = tid;
        float m = -INFINITY;
        #pragma unroll 4
        for (int j = 0; j < WIN; ++j) m = fmaxf(m, attn[row * ASTR + j]);
        float s = 0.0f;
        #pragma unroll 4
        for (int j = 0; j < WIN; ++j) {
            const float e = __expf(attn[row * ASTR + j] - m);
            attn[row * ASTR + j] = e;
            s += e;
        }
        const float rinv = 1.0f / s;
        #pragma unroll 4
        for (int j = 0; j < WIN; ++j)      // A[row][k], k = row + j (banded)
            wmat[row * WSTR + row + j] = (_Float16)(attn[row * ASTR + j] * rinv);
    }
    __syncthreads();

    // ---------------- phase 3: out(16x64) = W(16x96) x V(96x64) via WMMA ---
    const int wave  = tid >> 5;            // each wave owns 16 output columns
    const int lane  = tid & 31;
    const int arow  = lane & 15;           // A: M = lane % 16
    const int koffA = (lane < 16) ? 0 : 16;// A K-half select per half-wave
    const int koffB = (lane < 16) ? 0 : 16;// B K-half select per half-wave
    const int col   = lane & 15;           // B/D: N = lane % 16

    v8f c = {};
    #pragma unroll
    for (int kt = 0; kt < NKT; ++kt) {
        // A fragment: halves 0-15 = K koffA..+15 (contiguous)
        union { v16h v; v8h h[2]; } a;
        const _Float16* wr = &wmat[arow * WSTR + kt * 32 + koffA];
        a.h[0] = *reinterpret_cast<const v8h*>(wr);
        a.h[1] = *reinterpret_cast<const v8h*>(wr + 8);
        // B fragment: 16 contiguous K values of column (16*wave + col)
        const v16h b = *reinterpret_cast<const v16h*>(
            &vsT[(wave * 16 + col) * VSTR + kt * 32 + koffB]);
        c = __builtin_amdgcn_wmma_f32_16x16x32_f16(
                false, a.v, false, b, (short)0, c, false, false);
    }

    // D layout: VGPR r -> M = r (lanes 0-15) / r+8 (lanes 16-31), N = lane%16
    const int N     = wave * 16 + col;
    const int mbase = (lane < 16) ? 0 : 8;
    #pragma unroll
    for (int r = 0; r < 8; ++r) {
        const int tokd = d0 + mbase + r;
        out[(((size_t)bb * NTOK + tokd) * NH + hh) * WID + N] = c[r];
    }
}

extern "C" void kernel_launch(void* const* d_in, const int* in_sizes, int n_in,
                              void* d_out, int out_size, void* d_ws, size_t ws_size,
                              hipStream_t stream) {
    // setup_inputs order: v, q, k, coo, dst_mxlen, src_mxlen
    const float* v = (const float*)d_in[0];
    const float* q = (const float*)d_in[1];
    const float* k = (const float*)d_in[2];
    float* out = (float*)d_out;

    const int blocks = 2 /*bs*/ * NH * (NTOK / DT);   // 1024
    l1attn_win_kernel<<<dim3(blocks), dim3(128), 0, stream>>>(v, q, k, out);
}
